// QLSTM_65481071396875
// MI455X (gfx1250) — matrix operand order repaired and checked by reading, not compiled, not run on hardware
//
#include <hip/hip_runtime.h>
#include <hip/hip_fp16.h>

// ---------------------------------------------------------------------------
// Persistent fused LSTM for MI455X (gfx1250, wave32, WMMA).
//   T=2048, B=64, D=256, H=256.  Latency-bound recurrent chain -> persistent
//   kernel: weights resident in LDS as pre-swizzled f16 WMMA B-fragments,
//   cell state c resident in VGPRs, one grid barrier per timestep with
//   double-buffered f16 h state, x_{t+1} prefetched during step t.
// ---------------------------------------------------------------------------

typedef __attribute__((ext_vector_type(16))) _Float16 v16h;
typedef __attribute__((ext_vector_type(8)))  _Float16 v8h;
typedef __attribute__((ext_vector_type(4)))  _Float16 v4h;
typedef __attribute__((ext_vector_type(8)))  float    v8f;

#define T_STEPS   2048
#define BATCH     64
#define DIMD      256
#define DIMH      256
#define DHSUM     512            // D + H (GEMM K dimension)
#define NWG       16             // workgroups; each owns 16 hidden columns
#define NTHREADS  128            // 4 wave32 waves; wave w = batch M-tile w
#define KT        (DHSUM / 32)   // 16 k-tiles of 32

// LDS layout (halves):
//   comb : BATCH rows x COMB_STRIDE halves  ([x_t | h_{t-1}] as f16)
//   wfrag: KT * 4gates * 32lanes * 16 halves (WMMA B-fragment order)
#define COMB_STRIDE  520                         // 512 + 8 pad; row = 1040B (16B aligned)
#define COMB_HALVES  (BATCH * COMB_STRIDE)       // 33280
#define WFRAG_TOTAL  (KT * 4 * 32 * 16)          // 32768
// total LDS = (33280 + 32768) * 2 = 132096 bytes (~129 KB) of the 320 KB WGP pool

__global__ __launch_bounds__(NTHREADS)
void qlstm_persistent(const float* __restrict__ x,
                      const float* __restrict__ Wf, const float* __restrict__ bf,
                      const float* __restrict__ Wi, const float* __restrict__ bi,
                      const float* __restrict__ Wgm, const float* __restrict__ bg,
                      const float* __restrict__ Wo, const float* __restrict__ bo,
                      float*    __restrict__ out,          // [T,B,H] ++ hx[B,H] ++ cx[B,H]
                      _Float16* __restrict__ hbuf,         // 2 * B*H f16 (ping-pong h state)
                      int*      __restrict__ barrier_cnt)  // 1 int, zeroed per call
{
    __shared__ __align__(16) _Float16 comb [COMB_HALVES];
    __shared__ __align__(16) _Float16 wfrag[WFRAG_TOTAL];

    const int wg   = blockIdx.x;            // owns hidden cols [wg*16, wg*16+16)
    const int tid  = threadIdx.x;
    const int wave = tid >> 5;              // 0..3 : batch M-tile
    const int lane = tid & 31;
    const int hlf  = lane >> 4;             // 0/1  : K-half select (A/B frag layout)
    const int ln16 = lane & 15;

    // ---- one-time: stage this WG's weight slice into LDS, B-fragment order ----
    // wfrag[((kt*4+g)*32 + l)*16 + e] = f16( W_g[ kt*32 + (e>>3)*16 + (l>>4)*8 + (e&7) ][ wg*16 + (l&15) ] )
    for (int idx = tid; idx < WFRAG_TOTAL; idx += NTHREADS) {
        int e  = idx & 15;
        int l  = (idx >> 4) & 31;
        int g  = (idx >> 9) & 3;
        int kt = idx >> 11;
        const float* Wp = (g == 0) ? Wf : (g == 1) ? Wi : (g == 2) ? Wgm : Wo;
        int krow = kt * 32 + ((e >> 3) << 4) + ((l >> 4) << 3) + (e & 7);
        int col  = (wg << 4) + (l & 15);
        wfrag[idx] = (_Float16)Wp[krow * DIMH + col];
    }

    // per-lane output column + biases
    const int   jcol = (wg << 4) + ln16;
    const float bfv = bf[jcol], biv = bi[jcol], bgv = bg[jcol], bov = bo[jcol];

    __syncthreads();

    float creg[8];   // resident cell state: rows (wave*16 + hlf*8 + r), col jcol
    float hl[8];
    #pragma unroll
    for (int r = 0; r < 8; ++r) { creg[r] = 0.0f; hl[r] = 0.0f; }

    const v8f vzero = {0.f, 0.f, 0.f, 0.f, 0.f, 0.f, 0.f, 0.f};
    const int arow = (wave << 4) + ln16;    // A-fragment batch row for this lane

    for (int t = 0; t < T_STEPS; ++t) {
        // ---- grid barrier: all WGs finished step t-1 (reads AND writes) ----
        if (t > 0) {
            __threadfence();                 // release my stores to device scope
            __syncthreads();
            if (tid == 0) {
                atomicAdd(barrier_cnt, 1);
                const int target = t * NWG;
                while (__hip_atomic_load(barrier_cnt, __ATOMIC_ACQUIRE,
                                         __HIP_MEMORY_SCOPE_AGENT) < target) {
                    __builtin_amdgcn_s_sleep(1);
                }
            }
            __syncthreads();
        }

        // ---- stage x_t -> comb[:, 0:256] (f32 -> f16, packed 8B stores) ----
        const float* xt = x + (size_t)t * BATCH * DIMD;
        for (int idx = tid; idx < BATCH * (DIMD / 4); idx += NTHREADS) {
            int row = idx >> 6;              // 64 float4 groups per row
            int c4  = (idx & 63) << 2;       // 0..252
            float4 v = *(const float4*)(xt + row * DIMD + c4);
            v4h p;
            p[0] = (_Float16)v.x; p[1] = (_Float16)v.y;
            p[2] = (_Float16)v.z; p[3] = (_Float16)v.w;
            *(v4h*)(comb + row * COMB_STRIDE + c4) = p;
        }

        // ---- stage h_{t-1} -> comb[:, 256:512] (f16 straight copy, 16B) ----
        const _Float16* hprv = hbuf + (size_t)(t & 1) * BATCH * DIMH;
        for (int idx = tid; idx < BATCH * (DIMH / 8); idx += NTHREADS) {
            int row = idx >> 5;              // 32 v8h chunks per row
            int c8  = (idx & 31) << 3;       // 0..248
            v8h v = *(const v8h*)(hprv + row * DIMH + c8);
            *(v8h*)(comb + row * COMB_STRIDE + DIMD + c8) = v;
        }

        // ---- prefetch next timestep's x slice (global_prefetch_b8) ----
        if (t + 1 < T_STEPS) {
            const char* nx = (const char*)(x + (size_t)(t + 1) * BATCH * DIMD);
            #pragma unroll
            for (int p = 0; p < 4; ++p)
                __builtin_prefetch(nx + (size_t)(tid + p * NTHREADS) * 128, 0, 1);
        }
        __syncthreads();

        // ---- GEMM: [16 batch rows, 512] x [512, 16 cols] for all 4 gates ----
        v8f acc[4];
        #pragma unroll
        for (int g = 0; g < 4; ++g) acc[g] = vzero;

        #pragma unroll 4
        for (int kt = 0; kt < KT; ++kt) {
            // A fragment (16x32 f16): lane holds row arow,
            //   K = kt*32 + hlf*8 + [0..7]  and  kt*32 + 16 + hlf*8 + [0..7]
            union { v16h v; v8h h[2]; } a;
            const _Float16* abase = comb + arow * COMB_STRIDE + kt * 32 + (hlf << 3);
            a.h[0] = *(const v8h*)(abase);
            a.h[1] = *(const v8h*)(abase + 16);

            const _Float16* wb = wfrag + (((kt * 4) * 32 + lane) << 4);
            #pragma unroll
            for (int g = 0; g < 4; ++g) {
                v16h bfr = *(const v16h*)(wb + ((g * 32) << 4));
                acc[g] = __builtin_amdgcn_wmma_f32_16x16x32_f16(
                    false, a.v, false, bfr, (short)0, acc[g], false, false);
            }
        }

        // ---- elementwise cell update + writes ----
        _Float16* hnew = hbuf + (size_t)((t + 1) & 1) * BATCH * DIMH;
        #pragma unroll
        for (int r = 0; r < 8; ++r) {
            const int brow = (wave << 4) + (hlf << 3) + r;   // C/D layout: M = r + 8*hlf
            float fz = acc[0][r] + bfv;
            float iz = acc[1][r] + biv;
            float gz = acc[2][r] + bgv;
            float oz = acc[3][r] + bov;
            float fg = 1.0f / (1.0f + __expf(-fz));
            float ig = 1.0f / (1.0f + __expf(-iz));
            float gg = tanhf(gz);
            float og = 1.0f / (1.0f + __expf(-oz));
            float c  = fg * creg[r] + ig * gg;
            creg[r]  = c;
            float h  = og * tanhf(c);
            hl[r]    = h;
            out[(size_t)t * BATCH * DIMH + (size_t)brow * DIMH + jcol] = h;
            hnew[brow * DIMH + jcol] = (_Float16)h;
        }
    }

    // ---- final hx, cx (tuple tail of d_out) ----
    float* hx = out + (size_t)T_STEPS * BATCH * DIMH;
    float* cx = hx + BATCH * DIMH;
    #pragma unroll
    for (int r = 0; r < 8; ++r) {
        const int brow = (wave << 4) + (hlf << 3) + r;
        hx[brow * DIMH + jcol] = hl[r];
        cx[brow * DIMH + jcol] = creg[r];
    }
}

extern "C" void kernel_launch(void* const* d_in, const int* in_sizes, int n_in,
                              void* d_out, int out_size, void* d_ws, size_t ws_size,
                              hipStream_t stream) {
    (void)in_sizes; (void)n_in; (void)out_size; (void)ws_size;

    const float* x  = (const float*)d_in[0];
    const float* Wf = (const float*)d_in[1];
    const float* bf = (const float*)d_in[2];
    const float* Wi = (const float*)d_in[3];
    const float* bi = (const float*)d_in[4];
    const float* Wg = (const float*)d_in[5];
    const float* bg = (const float*)d_in[6];
    const float* Wo = (const float*)d_in[7];
    const float* bo = (const float*)d_in[8];

    float*    out  = (float*)d_out;
    _Float16* hbuf = (_Float16*)d_ws;                            // 2 * B*H f16
    int*      bar  = (int*)((char*)d_ws + 2 * BATCH * DIMH * sizeof(_Float16));

    // zero h0 ping-pong buffers + barrier counter (graph-capture safe, replayed)
    hipMemsetAsync(d_ws, 0, 2 * BATCH * DIMH * sizeof(_Float16) + 256, stream);

    hipLaunchKernelGGL(qlstm_persistent, dim3(NWG), dim3(NTHREADS), 0, stream,
                       x, Wf, bf, Wi, bi, Wg, bg, Wo, bo, out, hbuf, bar);
}
